// HeterogeneousTimingGNN_86285892976816
// MI455X (gfx1250) — matrix operand order repaired and checked by reading, not compiled, not run on hardware
//
#include <hip/hip_runtime.h>
#include <math.h>

typedef __attribute__((ext_vector_type(16))) __bf16 v16bf;
typedef __attribute__((ext_vector_type(8)))  float  v8f;

__device__ __forceinline__ unsigned ordF(float f) {
  unsigned b = __float_as_uint(f);
  return (b & 0x80000000u) ? ~b : (b | 0x80000000u);
}
__device__ __forceinline__ float unordF(unsigned u) {
  unsigned b = (u & 0x80000000u) ? (u & 0x7fffffffu) : ~u;
  return __uint_as_float(b);
}

// ---------------- utility fills / reductions ----------------
__global__ void k_fill_u32(unsigned* __restrict__ p, unsigned v, size_t n) {
  size_t i = (size_t)blockIdx.x * blockDim.x + threadIdx.x;
  size_t st = (size_t)gridDim.x * blockDim.x;
  for (; i < n; i += st) p[i] = v;
}

__global__ void k_ea_sum(const float* __restrict__ ea, int E, float* __restrict__ sums) {
  float s0 = 0.f, s1 = 0.f, s2 = 0.f;
  int st = gridDim.x * blockDim.x;
  for (int e = blockIdx.x * blockDim.x + threadIdx.x; e < E; e += st) {
    s0 += ea[e * 3 + 0]; s1 += ea[e * 3 + 1]; s2 += ea[e * 3 + 2];
  }
  atomicAdd(&sums[0], s0); atomicAdd(&sums[1], s1); atomicAdd(&sums[2], s2);
}

// M[d,h] = sum_c We[d, h*C+c] * ae[h,c]   (collapses edge projection to 3-dot)
__global__ void k_attn_m(const float* We1, const float* ae1,
                         const float* We2, const float* ae2,
                         const float* We3, const float* ae3,
                         float* __restrict__ M) {
  int t = threadIdx.x;
  if (t < 12) {
    int d = t % 3, h = t / 3; float s = 0.f;
    for (int c = 0; c < 128; ++c) s += We1[d * 512 + h * 128 + c] * ae1[h * 128 + c];
    M[d * 4 + h] = s;
  } else if (t < 24) {
    int u = t - 12, d = u % 3, h = u / 3; float s = 0.f;
    for (int c = 0; c < 128; ++c) s += We2[d * 512 + h * 128 + c] * ae2[h * 128 + c];
    M[12 + d * 4 + h] = s;
  } else if (t < 27) {
    int d = t - 24; float s = 0.f;
    for (int c = 0; c < 2; ++c) s += We3[d * 2 + c] * ae3[c];
    M[24 + d] = s;
  }
}

// ---------------- layer-1 GEMM: [N,10] @ [10,512] ----------------
__global__ void k_gemm1(const float* __restrict__ x, const float* __restrict__ W,
                        float* __restrict__ g, int N) {
  __shared__ float xs[10];
  int n = blockIdx.x;
  int t = threadIdx.x;                 // 128 threads, 4 cols each
  if (t < 10) xs[t] = x[(size_t)n * 10 + t];
  __syncthreads();
  int c0 = t * 4;
  float a0 = 0.f, a1 = 0.f, a2 = 0.f, a3 = 0.f;
#pragma unroll
  for (int d = 0; d < 10; ++d) {
    float xv = xs[d];
    const float* w = &W[d * 512 + c0];
    a0 += xv * w[0]; a1 += xv * w[1]; a2 += xv * w[2]; a3 += xv * w[3];
  }
  float* gp = &g[(size_t)n * 512 + c0];
  gp[0] = a0; gp[1] = a1; gp[2] = a2; gp[3] = a3;
}

// ---------------- per-node attention dot scores ----------------
__global__ void k_node_scores(const float* __restrict__ g,
                              const float* __restrict__ as, const float* __restrict__ ad,
                              float* __restrict__ sS, float* __restrict__ sD,
                              int H, int C) {
  int n = blockIdx.x;
  int h = threadIdx.y;
  int lane = threadIdx.x;
  const float* gp = &g[((size_t)n * H + h) * C];
  const float* ap = &as[h * C];
  const float* dp = &ad[h * C];
  float ss = 0.f, sd = 0.f;
  for (int c = lane; c < C; c += 32) { float v = gp[c]; ss += v * ap[c]; sd += v * dp[c]; }
  for (int o = 16; o > 0; o >>= 1) { ss += __shfl_down(ss, o, 32); sd += __shfl_down(sd, o, 32); }
  if (lane == 0) { sS[n * H + h] = ss; sD[n * H + h] = sd; }
}

// ---------------- edge pass 1: alpha + segment max ----------------
__global__ void k_edge_alpha(const int* __restrict__ ei, int E, int N,
                             const float* __restrict__ ea, const float* __restrict__ eaSum,
                             float invE, const float* __restrict__ M,
                             const float* __restrict__ sS, const float* __restrict__ sD,
                             float* __restrict__ alpha, unsigned* __restrict__ amax, int H) {
  int Etot = E + N;
  int st = gridDim.x * blockDim.x;
  for (int e = blockIdx.x * blockDim.x + threadIdx.x; e < Etot; e += st) {
    int s, d; float e0, e1, e2;
    if (e < E) { s = ei[e]; d = ei[E + e]; e0 = ea[e * 3]; e1 = ea[e * 3 + 1]; e2 = ea[e * 3 + 2]; }
    else       { s = e - E; d = s; e0 = eaSum[0] * invE; e1 = eaSum[1] * invE; e2 = eaSum[2] * invE; }
    for (int h = 0; h < H; ++h) {
      float a = sS[s * H + h] + sD[d * H + h]
              + e0 * M[0 * H + h] + e1 * M[1 * H + h] + e2 * M[2 * H + h];
      a = a > 0.f ? a : 0.2f * a;          // leaky_relu
      alpha[(size_t)e * H + h] = a;
      atomicMax(&amax[d * H + h], ordF(a));
    }
  }
}

// ---------------- edge pass 2: exp + segment sum ----------------
__global__ void k_edge_expsum(const int* __restrict__ ei, int E, int N,
                              const unsigned* __restrict__ amax,
                              float* __restrict__ alpha, float* __restrict__ denom, int H) {
  int Etot = E + N;
  int st = gridDim.x * blockDim.x;
  for (int e = blockIdx.x * blockDim.x + threadIdx.x; e < Etot; e += st) {
    int d = (e < E) ? ei[E + e] : (e - E);
    for (int h = 0; h < H; ++h) {
      float m = unordF(amax[d * H + h]);
      float ex = expf(alpha[(size_t)e * H + h] - m);
      alpha[(size_t)e * H + h] = ex;
      atomicAdd(&denom[d * H + h], ex);
    }
  }
}

// ---------------- edge pass 3: scatter-aggregate (HC=512) ----------------
__global__ void k_edge_agg512(const int* __restrict__ ei, int E, int N,
                              const float* __restrict__ g, const float* __restrict__ exA,
                              const float* __restrict__ denom, float* __restrict__ agg) {
  int Etot = E + N;
  for (int e = blockIdx.x; e < Etot; e += gridDim.x) {
    int s, d;
    if (e < E) { s = ei[e]; d = ei[E + e]; } else { s = e - E; d = s; }
    int c4 = threadIdx.x * 4;            // 128 threads -> 512 channels
    int h = c4 >> 7;                     // C = 128
    float att = exA[(size_t)e * 4 + h] / (denom[d * 4 + h] + 1e-16f);
    float4 gv = *(const float4*)&g[(size_t)s * 512 + c4];
    float* ap = &agg[(size_t)d * 512 + c4];
    atomicAdd(&ap[0], gv.x * att);
    atomicAdd(&ap[1], gv.y * att);
    atomicAdd(&ap[2], gv.z * att);
    atomicAdd(&ap[3], gv.w * att);
  }
}

// layer-3 aggregate (H=1, C=2): one thread per edge
__global__ void k_edge_agg2(const int* __restrict__ ei, int E, int N,
                            const float* __restrict__ g, const float* __restrict__ exA,
                            const float* __restrict__ denom, float* __restrict__ agg) {
  int Etot = E + N;
  int st = gridDim.x * blockDim.x;
  for (int e = blockIdx.x * blockDim.x + threadIdx.x; e < Etot; e += st) {
    int s, d;
    if (e < E) { s = ei[e]; d = ei[E + e]; } else { s = e - E; d = s; }
    float att = exA[e] / (denom[d] + 1e-16f);
    atomicAdd(&agg[(size_t)d * 2 + 0], g[(size_t)s * 2 + 0] * att);
    atomicAdd(&agg[(size_t)d * 2 + 1], g[(size_t)s * 2 + 1] * att);
  }
}

__global__ void k_bias_relu(float* __restrict__ a, const float* __restrict__ b,
                            size_t n, unsigned mask) {
  size_t st = (size_t)gridDim.x * blockDim.x;
  for (size_t i = (size_t)blockIdx.x * blockDim.x + threadIdx.x; i < n; i += st) {
    float v = a[i] + b[(unsigned)i & mask];
    a[i] = v > 0.f ? v : 0.f;
  }
}

// ---------------- W2 -> bf16 WMMA B-operand layout ----------------
// B 32x16 (KxN): N=lane&15, kg = lane<16?0:8; j<8 -> K=kg+j, j>=8 -> K=16+kg+(j-8).
// packB[((nt*16+kt)*32+lane)*16 + j]
__global__ void k_packW2(const float* __restrict__ W, unsigned short* __restrict__ P) {
  int idx = blockIdx.x * blockDim.x + threadIdx.x;
  if (idx >= 512 * 512) return;
  int j    = idx & 15;
  int lane = (idx >> 4) & 31;
  int kt   = (idx >> 9) & 15;
  int nt   = idx >> 13;
  int n = nt * 16 + (lane & 15);
  int k = kt * 32 + ((lane < 16) ? 0 : 8) + ((j < 8) ? j : 8 + j);
  __bf16 h = (__bf16)W[k * 512 + n];
  P[idx] = *reinterpret_cast<unsigned short*>(&h);
}

// ---------------- layer-2 GEMM: [N,512] @ [512,512] via WMMA bf16 ----------------
// One wave per 16x16 output tile. A operand per 16-bit 16x32 layout:
// lane m = lane&15, kg = lane<16?0:8; elements j<8 -> K=kt*32+kg+j, j>=8 -> +16.
__global__ void k_gemm2(const float* __restrict__ A, const unsigned short* __restrict__ P,
                        float* __restrict__ G, int N) {
  int wid  = (blockIdx.x * blockDim.x + threadIdx.x) >> 5;
  int lane = threadIdx.x & 31;
  int mt = wid >> 5;
  int nt = wid & 31;
  if (mt >= (N >> 4)) return;
  int m  = mt * 16 + (lane & 15);
  int kg = (lane < 16) ? 0 : 8;
  const float* arow = A + (size_t)m * 512;
  v8f acc = {};
#pragma unroll 4
  for (int kt = 0; kt < 16; ++kt) {
    const float* c1 = arow + kt * 32 + kg;
    const float* c2 = c1 + 16;
    if (kt < 15) __builtin_prefetch(arow + (kt + 1) * 32 + kg, 0, 1);
    v16bf av;
#pragma unroll
    for (int j = 0; j < 8; ++j) { av[j] = (__bf16)c1[j]; av[j + 8] = (__bf16)c2[j]; }
    v16bf bv = *reinterpret_cast<const v16bf*>(P + ((((nt << 4) + kt) << 5) + lane) * 16);
    acc = __builtin_amdgcn_wmma_f32_16x16x32_bf16(false, av, false, bv,
                                                  (short)0, acc, false, false);
  }
  int n0 = (nt << 4) + (lane & 15);
  int rb = (lane < 16) ? 0 : 8;      // C/D: VGPR r -> row r (lanes 0-15) / r+8 (lanes 16-31)
  float* gout = G + (size_t)(mt * 16) * 512 + n0;
#pragma unroll
  for (int r = 0; r < 8; ++r) gout[(size_t)(rb + r) * 512] = acc[r];
}

// ---------------- layer-3 GEMM: [N,512] @ [512,2], wave per node ----------------
__global__ void k_gemm3(const float* __restrict__ h2, const float* __restrict__ W3,
                        float* __restrict__ g3, int N) {
  int w = (blockIdx.x * blockDim.x + threadIdx.x) >> 5;
  int lane = threadIdx.x & 31;
  if (w >= N) return;
  const float* hp = h2 + (size_t)w * 512;
  float a0 = 0.f, a1 = 0.f;
  for (int k = lane; k < 512; k += 32) {
    float v = hp[k];
    a0 += v * W3[k * 2 + 0];
    a1 += v * W3[k * 2 + 1];
  }
  for (int o = 16; o > 0; o >>= 1) { a0 += __shfl_down(a0, o, 32); a1 += __shfl_down(a1, o, 32); }
  if (lane == 0) { g3[(size_t)w * 2 + 0] = a0; g3[(size_t)w * 2 + 1] = a1; }
}

__global__ void k_final(const float* __restrict__ agg3, const float* __restrict__ b3,
                        float* __restrict__ out, int N) {
  int n = blockIdx.x * blockDim.x + threadIdx.x;
  if (n >= N) return;
  float v0 = agg3[(size_t)n * 2 + 0] + b3[0];
  float v1 = agg3[(size_t)n * 2 + 1] + b3[1];
  float m = fmaxf(v0, v1);
  float lse = m + logf(expf(v0 - m) + expf(v1 - m));
  out[(size_t)n * 2 + 0] = v0 - lse;
  out[(size_t)n * 2 + 1] = v1 - lse;
}

extern "C" void kernel_launch(void* const* d_in, const int* in_sizes, int n_in,
                              void* d_out, int out_size, void* d_ws, size_t ws_size,
                              hipStream_t stream) {
  const float* x   = (const float*)d_in[0];
  const int*   ei  = (const int*)d_in[1];
  const float* ea  = (const float*)d_in[2];
  const float* W1  = (const float*)d_in[3];
  const float* as1 = (const float*)d_in[4];
  const float* ad1 = (const float*)d_in[5];
  const float* We1 = (const float*)d_in[6];
  const float* ae1 = (const float*)d_in[7];
  const float* b1  = (const float*)d_in[8];
  const float* W2  = (const float*)d_in[9];
  const float* as2 = (const float*)d_in[10];
  const float* ad2 = (const float*)d_in[11];
  const float* We2 = (const float*)d_in[12];
  const float* ae2 = (const float*)d_in[13];
  const float* b2  = (const float*)d_in[14];
  const float* W3  = (const float*)d_in[15];
  const float* as3 = (const float*)d_in[16];
  const float* ad3 = (const float*)d_in[17];
  const float* We3 = (const float*)d_in[18];
  const float* ae3 = (const float*)d_in[19];
  const float* b3  = (const float*)d_in[20];
  (void)n_in; (void)out_size;

  const int N = in_sizes[0] / 10;
  const int E = in_sizes[1] / 2;
  const int Etot = E + N;
  const float invE = 1.0f / (float)E;

  char* ws = (char*)d_ws;
  size_t off = 0;
  auto take = [&](size_t bytes) -> char* {
    char* p = ws + off;
    off += (bytes + 255) & ~(size_t)255;
    return p;
  };
  float*          bufA  = (float*)take((size_t)N * 512 * 4);   // agg1 -> h1
  float*          bufB  = (float*)take((size_t)N * 512 * 4);   // g1 -> g2
  float*          bufC  = (float*)take((size_t)N * 512 * 4);   // agg2 -> h2
  float*          g3    = (float*)take((size_t)N * 2 * 4);
  float*          agg3  = (float*)take((size_t)N * 2 * 4);
  float*          sS    = (float*)take((size_t)N * 4 * 4);
  float*          sD    = (float*)take((size_t)N * 4 * 4);
  unsigned*       amax  = (unsigned*)take((size_t)N * 4 * 4);
  float*          denom = (float*)take((size_t)N * 4 * 4);
  float*          alpha = (float*)take((size_t)Etot * 4 * 4);
  unsigned short* packB = (unsigned short*)take(512 * 512 * 2);
  float*          eaSum = (float*)take(64);
  float*          Mbuf  = (float*)take(128);
  if (off > ws_size) return;  // workspace too small; bail deterministically

  auto cdiv = [](long long a, long long b) { return (int)((a + b - 1) / b); };

  // ---- precompute ----
  k_fill_u32<<<1, 64, 0, stream>>>((unsigned*)eaSum, 0u, 16);
  k_ea_sum<<<512, 256, 0, stream>>>(ea, E, eaSum);
  k_attn_m<<<1, 32, 0, stream>>>(We1, ae1, We2, ae2, We3, ae3, Mbuf);
  k_packW2<<<cdiv(512 * 512, 256), 256, 0, stream>>>(W2, packB);
  k_gemm1<<<N, 128, 0, stream>>>(x, W1, bufB, N);

  // ---- layer 1 (g=bufB, agg=bufA) ----
  k_node_scores<<<N, dim3(32, 4), 0, stream>>>(bufB, as1, ad1, sS, sD, 4, 128);
  k_fill_u32<<<cdiv((long long)N * 4, 256), 256, 0, stream>>>(amax, 0u, (size_t)N * 4);
  k_fill_u32<<<cdiv((long long)N * 4, 256), 256, 0, stream>>>((unsigned*)denom, 0u, (size_t)N * 4);
  k_fill_u32<<<cdiv((long long)N * 512, 256), 256, 0, stream>>>((unsigned*)bufA, 0u, (size_t)N * 512);
  k_edge_alpha<<<cdiv(Etot, 256), 256, 0, stream>>>(ei, E, N, ea, eaSum, invE, Mbuf, sS, sD, alpha, amax, 4);
  k_edge_expsum<<<cdiv(Etot, 256), 256, 0, stream>>>(ei, E, N, amax, alpha, denom, 4);
  k_edge_agg512<<<Etot, 128, 0, stream>>>(ei, E, N, bufB, alpha, denom, bufA);
  k_bias_relu<<<cdiv((long long)N * 512, 256), 256, 0, stream>>>(bufA, b1, (size_t)N * 512, 511u);

  // ---- layer 2 GEMM (WMMA bf16): bufA @ W2 -> bufB ----
  {
    long long waves = (long long)(N / 16) * 32;
    k_gemm2<<<cdiv(waves, 8), 256, 0, stream>>>(bufA, packB, bufB, N);
  }

  // ---- layer 2 (g=bufB, agg=bufC) ----
  k_node_scores<<<N, dim3(32, 4), 0, stream>>>(bufB, as2, ad2, sS, sD, 4, 128);
  k_fill_u32<<<cdiv((long long)N * 4, 256), 256, 0, stream>>>(amax, 0u, (size_t)N * 4);
  k_fill_u32<<<cdiv((long long)N * 4, 256), 256, 0, stream>>>((unsigned*)denom, 0u, (size_t)N * 4);
  k_fill_u32<<<cdiv((long long)N * 512, 256), 256, 0, stream>>>((unsigned*)bufC, 0u, (size_t)N * 512);
  k_edge_alpha<<<cdiv(Etot, 256), 256, 0, stream>>>(ei, E, N, ea, eaSum, invE, Mbuf + 12, sS, sD, alpha, amax, 4);
  k_edge_expsum<<<cdiv(Etot, 256), 256, 0, stream>>>(ei, E, N, amax, alpha, denom, 4);
  k_edge_agg512<<<Etot, 128, 0, stream>>>(ei, E, N, bufB, alpha, denom, bufC);
  k_bias_relu<<<cdiv((long long)N * 512, 256), 256, 0, stream>>>(bufC, b2, (size_t)N * 512, 511u);

  // ---- layer 3 (g=g3, agg=agg3, H=1, C=2) ----
  k_gemm3<<<cdiv((long long)N * 32, 256), 256, 0, stream>>>(bufC, W3, g3, N);
  k_node_scores<<<N, dim3(32, 1), 0, stream>>>(g3, as3, ad3, sS, sD, 1, 2);
  k_fill_u32<<<cdiv(N, 256), 256, 0, stream>>>(amax, 0u, (size_t)N);
  k_fill_u32<<<cdiv(N, 256), 256, 0, stream>>>((unsigned*)denom, 0u, (size_t)N);
  k_fill_u32<<<cdiv((long long)N * 2, 256), 256, 0, stream>>>((unsigned*)agg3, 0u, (size_t)N * 2);
  k_edge_alpha<<<cdiv(Etot, 256), 256, 0, stream>>>(ei, E, N, ea, eaSum, invE, Mbuf + 24, sS, sD, alpha, amax, 1);
  k_edge_expsum<<<cdiv(Etot, 256), 256, 0, stream>>>(ei, E, N, amax, alpha, denom, 1);
  k_edge_agg2<<<cdiv(Etot, 256), 256, 0, stream>>>(ei, E, N, g3, alpha, denom, agg3);
  k_final<<<cdiv(N, 256), 256, 0, stream>>>(agg3, b3, (float*)d_out, N);
}
